// SelfAttention_27908697490137
// MI455X (gfx1250) — matrix operand order repaired and checked by reading, not compiled, not run on hardware
//
#include <hip/hip_runtime.h>
#include <stdint.h>

// ---------------------------------------------------------------------------
// SAGAN self-attention, fused flash-style, bf16 WMMA + async-LDS (gfx1250).
// B=8, N=64*64=4096, C=512, d=C/8=64.
// ---------------------------------------------------------------------------

#define BB 8
#define NN 4096
#define CC 512
#define DD 64
#define HPAD 40   // padded row stride (elems) for LDS tiles: 80B -> conflict-free b128

typedef __attribute__((ext_vector_type(16))) __bf16 v16bf;
typedef __attribute__((ext_vector_type(8)))  float  v8f;

union AF { v16bf v; uint4 q[2]; unsigned short u[16]; };
union CF { v8f v; float f[8]; };

__device__ __forceinline__ unsigned short f2bf(float f) {
  unsigned u = __float_as_uint(f);
  u += 0x7FFFu + ((u >> 16) & 1u);          // round-to-nearest-even
  return (unsigned short)(u >> 16);
}

__device__ __forceinline__ v8f wmma_bf16(const AF& a, const AF& b, v8f c) {
  return __builtin_amdgcn_wmma_f32_16x16x32_bf16(false, a.v, false, b.v,
                                                 (short)0, c, false, false);
}

// ----------------------- async global->LDS helpers -------------------------

#if defined(__has_builtin)
#if __has_builtin(__builtin_amdgcn_global_load_async_to_lds_b128)
#define USE_ASYNC_LDS 1
#endif
#if __has_builtin(__builtin_amdgcn_s_wait_asynccnt)
#define HAVE_WAIT_ASYNC 1
#endif
#endif

typedef int vsi4 __attribute__((vector_size(16)));  // matches builtin param type

template <int OFF>
__device__ __forceinline__ void async_copy16(const unsigned short* gsrc,
                                             unsigned short* ldst) {
#ifdef USE_ASYNC_LDS
  __builtin_amdgcn_global_load_async_to_lds_b128(
      (__attribute__((address_space(1))) vsi4*)gsrc,
      (__attribute__((address_space(3))) vsi4*)ldst, OFF, 0);
#else
  *(uint4*)((char*)ldst + OFF) = *(const uint4*)((const char*)gsrc + OFF);
#endif
}

__device__ __forceinline__ void wait_async0() {
#if defined(HAVE_WAIT_ASYNC)
  __builtin_amdgcn_s_wait_asynccnt(0);
#elif defined(USE_ASYNC_LDS)
  asm volatile("s_wait_asynccnt 0" ::: "memory");
#endif
}

// --------------------------- conversion kernels ----------------------------

__global__ void k_cvt_x(const float* __restrict__ x,
                        unsigned short* __restrict__ xb, int n4) {
  int i = blockIdx.x * blockDim.x + threadIdx.x;
  if (i >= n4) return;
  float4 f = ((const float4*)x)[i];
  ushort4 o;
  o.x = f2bf(f.x); o.y = f2bf(f.y); o.z = f2bf(f.z); o.w = f2bf(f.w);
  ((ushort4*)xb)[i] = o;
}

// w: [K][Nc] fp32 row-major -> wT: [Nc][K] bf16 (so B-fragments are contiguous)
__global__ void k_cvt_wT(const float* __restrict__ w,
                         unsigned short* __restrict__ wT, int K, int Nc) {
  int i = blockIdx.x * blockDim.x + threadIdx.x;
  if (i >= K * Nc) return;
  int k = i / Nc, n = i % Nc;
  wT[(size_t)n * K + k] = f2bf(w[i]);
}

// --------------------------- projection GEMM -------------------------------
// out[M x Nc] = A[M x K] @ W[K x Nc] + bias, per batch; A,W,out in bf16.
// W passed transposed [Nc][K]. transposeOut=1 stores out as [Nc][M] (for hT).
__global__ __launch_bounds__(128)
void k_proj(const unsigned short* __restrict__ A,
            const unsigned short* __restrict__ WT,
            const float* __restrict__ bias,
            unsigned short* __restrict__ out,
            int M, int K, int Nc, int transposeOut) {
  const int wave = threadIdx.x >> 5;
  const int lane = threadIdx.x & 31;
  const int laneN = lane & 15, half = lane >> 4;
  const int m0 = (blockIdx.x * 4 + wave) * 16;
  const int n0 = blockIdx.y * 16;
  const int bz = blockIdx.z;

  const unsigned short* Ab = A + (size_t)bz * M * K;
  const unsigned short* Ar = Ab + (size_t)(m0 + laneN) * K + half * 16;
  const unsigned short* Br = WT + (size_t)(n0 + laneN) * K + half * 16;

  CF acc = {};
  for (int kk = 0; kk < K; kk += 32) {
    AF a, b;
    a.q[0] = *(const uint4*)(Ar + kk);
    a.q[1] = *(const uint4*)(Ar + kk + 8);
    b.q[0] = *(const uint4*)(Br + kk);
    b.q[1] = *(const uint4*)(Br + kk + 8);
    acc.v = wmma_bf16(a, b, acc.v);
  }

  const float bv = bias[n0 + laneN];
#pragma unroll
  for (int v = 0; v < 8; ++v) {
    const int row = m0 + v + 8 * half;
    const int col = n0 + laneN;
    const unsigned short r = f2bf(acc.f[v] + bv);
    if (!transposeOut)
      out[(size_t)bz * M * Nc + (size_t)row * Nc + col] = r;
    else
      out[(size_t)bz * Nc * M + (size_t)col * M + row] = r;
  }
}

// --------------------------- fused attention -------------------------------
// WG = 512 threads = 16 waves covering 64 query rows x 512 channels.
// wave (qi, ci): qi = wave>>2 -> 16-row query sub-block; ci = wave&3 -> 128 ch.
// Per 32-key step: h tile [32 x 512] async-staged to LDS (double-buffered,
// shared by all 16 waves); the 4 ci==0 waves compute s + online softmax once
// and broadcast p / rescale factors via LDS; all waves run 8 o-WMMAs.
__global__ __launch_bounds__(512)
void k_attn(const unsigned short* __restrict__ g,   // [B][N][64] bf16
            const unsigned short* __restrict__ f,   // [B][N][64] bf16
            const unsigned short* __restrict__ hT,  // [B][C][N] bf16
            const float* __restrict__ x,            // [B][N][C] fp32
            const float* __restrict__ gamma_p,
            float* __restrict__ y) {                // [B][N][C] fp32
  __shared__ __align__(16) unsigned short hbuf[2][CC * HPAD];  // 2 x 40 KB
  __shared__ __align__(16) unsigned short pbuf[4][16 * HPAD];  // 5 KB
  __shared__ __align__(16) float sbuf[4][16];                  // rescale bcast
  __shared__ __align__(16) float rbuf[4][16];                  // rowsum bcast

  const int tid = threadIdx.x;
  const int wave = tid >> 5;
  const int lane = tid & 31;
  const int laneN = lane & 15, half = lane >> 4;
  const int qi = wave >> 2;                 // 0..3 query sub-block
  const int ci = wave & 3;                  // 0..3 channel group
  const int q0 = blockIdx.x * 64 + qi * 16;
  const int bi = blockIdx.y;
  const int c0 = ci * 128;

  const unsigned short* gB = g + (size_t)bi * NN * DD;
  const unsigned short* fB = f + (size_t)bi * NN * DD;
  const unsigned short* hB = hT + (size_t)bi * CC * NN;

  // staging: thread t owns channel t -> 64B (32 keys) per step
  const unsigned short* hcol = hB + (size_t)tid * NN;
  unsigned short* lcol[2] = { &hbuf[0][tid * HPAD], &hbuf[1][tid * HPAD] };

  // A-fragments of G (only ci==0 waves use them)
  AF aG0, aG1;
  if (ci == 0) {
    const unsigned short* gr = gB + (size_t)(q0 + laneN) * DD + half * 16;
    aG0.q[0] = *(const uint4*)(gr);
    aG0.q[1] = *(const uint4*)(gr + 8);
    aG1.q[0] = *(const uint4*)(gr + 32);
    aG1.q[1] = *(const uint4*)(gr + 40);
  }

  CF acc[8];
#pragma unroll
  for (int c = 0; c < 8; ++c) acc[c] = CF{};
  float rmax[8], rsum[8];
#pragma unroll
  for (int v = 0; v < 8; ++v) { rmax[v] = -3.0e38f; rsum[v] = 0.0f; }

  // prefetch first h tile
  {
    const unsigned short* gsrc = hcol;
    async_copy16<0>(gsrc, lcol[0]);
    async_copy16<16>(gsrc, lcol[0]);
    async_copy16<32>(gsrc, lcol[0]);
    async_copy16<48>(gsrc, lcol[0]);
  }

#pragma unroll 1
  for (int mb = 0; mb < NN; mb += 32) {
    const int buf = (mb >> 5) & 1;
    wait_async0();
    __syncthreads();   // h tile[buf] visible; prior-step LDS reads retired

    if (mb + 32 < NN) {  // prefetch next tile into the other buffer
      const unsigned short* gsrc = hcol + mb + 32;
      unsigned short* ldst = lcol[buf ^ 1];
      async_copy16<0>(gsrc, ldst);
      async_copy16<16>(gsrc, ldst);
      async_copy16<32>(gsrc, ldst);
      async_copy16<48>(gsrc, ldst);
    }

    if (ci == 0) {
      // ---- S tiles: s[q, mb..mb+15], s[q, mb+16..mb+31] = G @ F^T ----
      CF s0 = {}, s1 = {};
      {
        AF b;
        const unsigned short* fr0 = fB + (size_t)(mb + laneN) * DD + half * 16;
        b.q[0] = *(const uint4*)(fr0);
        b.q[1] = *(const uint4*)(fr0 + 8);
        s0.v = wmma_bf16(aG0, b, s0.v);
        b.q[0] = *(const uint4*)(fr0 + 32);
        b.q[1] = *(const uint4*)(fr0 + 40);
        s0.v = wmma_bf16(aG1, b, s0.v);
        const unsigned short* fr1 =
            fB + (size_t)(mb + 16 + laneN) * DD + half * 16;
        b.q[0] = *(const uint4*)(fr1);
        b.q[1] = *(const uint4*)(fr1 + 8);
        s1.v = wmma_bf16(aG0, b, s1.v);
        b.q[0] = *(const uint4*)(fr1 + 32);
        b.q[1] = *(const uint4*)(fr1 + 40);
        s1.v = wmma_bf16(aG1, b, s1.v);
      }
      // ---- online softmax; row = v + 8*half spans the 16-lane half ----
      unsigned short* pw = pbuf[qi];
#pragma unroll
      for (int v = 0; v < 8; ++v) {
        float x0 = s0.f[v], x1 = s1.f[v];
        float tm = fmaxf(x0, x1);
#pragma unroll
        for (int off = 1; off < 16; off <<= 1)
          tm = fmaxf(tm, __shfl_xor(tm, off, 32));
        const float nm = fmaxf(rmax[v], tm);
        const float sc = __expf(rmax[v] - nm);
        rmax[v] = nm;
        const float p0 = __expf(x0 - nm);
        const float p1 = __expf(x1 - nm);
        float ps = p0 + p1;
#pragma unroll
        for (int off = 1; off < 16; off <<= 1)
          ps += __shfl_xor(ps, off, 32);
        rsum[v] = rsum[v] * sc + ps;
        const int row = v + 8 * half;
        pw[row * HPAD + laneN]      = f2bf(p0);
        pw[row * HPAD + 16 + laneN] = f2bf(p1);
        sbuf[qi][row] = sc;   // broadcast rescale factor
      }
    }
    __syncthreads();   // p / scale published; h tile stays valid

    // ---- rescale accumulators (scale read vectorized from LDS) ----
    {
      const float4 sA = *(const float4*)&sbuf[qi][8 * half];
      const float4 sB = *(const float4*)&sbuf[qi][8 * half + 4];
      const float sc[8] = {sA.x, sA.y, sA.z, sA.w, sB.x, sB.y, sB.z, sB.w};
#pragma unroll
      for (int c = 0; c < 8; ++c)
#pragma unroll
        for (int v = 0; v < 8; ++v) acc[c].f[v] *= sc[v];
    }

    // ---- reload P as an A-fragment (16x32, K = key offset) ----
    AF pA;
    {
      const unsigned short* lp = pbuf[qi] + laneN * HPAD + half * 16;
      pA.q[0] = *(const uint4*)(lp);
      pA.q[1] = *(const uint4*)(lp + 8);
    }

    // ---- o[16 x 128] += P[16 x 32] @ H[32 x 128] from LDS h tile ----
    const unsigned short* hbase = &hbuf[buf][0];
#pragma unroll
    for (int c = 0; c < 8; ++c) {
      AF hb;
      const unsigned short* hr =
          hbase + (c0 + c * 16 + laneN) * HPAD + half * 16;
      hb.q[0] = *(const uint4*)(hr);
      hb.q[1] = *(const uint4*)(hr + 8);
      acc[c].v = wmma_bf16(pA, hb, acc[c].v);
    }
  }

  // ---- publish row sums, then epilogue: y = gamma * o / rowsum + x ----
  if (ci == 0) {
#pragma unroll
    for (int v = 0; v < 8; ++v) rbuf[qi][v + 8 * half] = rsum[v];
  }
  __syncthreads();
  const float gm = gamma_p[0];
  const float4 rA = *(const float4*)&rbuf[qi][8 * half];
  const float4 rB = *(const float4*)&rbuf[qi][8 * half + 4];
  const float rs[8] = {rA.x, rA.y, rA.z, rA.w, rB.x, rB.y, rB.z, rB.w};
#pragma unroll
  for (int c = 0; c < 8; ++c) {
#pragma unroll
    for (int v = 0; v < 8; ++v) {
      const int row = q0 + v + 8 * half;
      const int col = c0 + c * 16 + laneN;
      const size_t idx = ((size_t)bi * NN + row) * CC + col;
      y[idx] = gm * (acc[c].f[v] / rs[v]) + x[idx];
    }
  }
}

// ------------------------------ launcher -----------------------------------

extern "C" void kernel_launch(void* const* d_in, const int* in_sizes, int n_in,
                              void* d_out, int out_size, void* d_ws, size_t ws_size,
                              hipStream_t stream) {
  const float* x     = (const float*)d_in[0];
  const float* wf    = (const float*)d_in[1];
  const float* bf_   = (const float*)d_in[2];
  const float* wg    = (const float*)d_in[3];
  const float* bg    = (const float*)d_in[4];
  const float* wh    = (const float*)d_in[5];
  const float* bh    = (const float*)d_in[6];
  const float* gamma = (const float*)d_in[7];
  float* y = (float*)d_out;

  // workspace layout (bf16 buffers), total ~76 MB
  char* ws = (char*)d_ws;
  size_t off = 0;
  unsigned short* xb  = (unsigned short*)(ws + off); off += (size_t)BB * NN * CC * 2;
  unsigned short* fb  = (unsigned short*)(ws + off); off += (size_t)BB * NN * DD * 2;
  unsigned short* gb  = (unsigned short*)(ws + off); off += (size_t)BB * NN * DD * 2;
  unsigned short* hTb = (unsigned short*)(ws + off); off += (size_t)BB * CC * NN * 2;
  unsigned short* wfT = (unsigned short*)(ws + off); off += (size_t)DD * CC * 2;
  unsigned short* wgT = (unsigned short*)(ws + off); off += (size_t)DD * CC * 2;
  unsigned short* whT = (unsigned short*)(ws + off); off += (size_t)CC * CC * 2;
  (void)off; (void)ws_size; (void)in_sizes; (void)n_in; (void)out_size;

  const int n4 = BB * NN * CC / 4;
  k_cvt_x<<<(n4 + 255) / 256, 256, 0, stream>>>(x, xb, n4);
  k_cvt_wT<<<(CC * DD + 255) / 256, 256, 0, stream>>>(wf, wfT, CC, DD);
  k_cvt_wT<<<(CC * DD + 255) / 256, 256, 0, stream>>>(wg, wgT, CC, DD);
  k_cvt_wT<<<(CC * CC + 255) / 256, 256, 0, stream>>>(wh, whT, CC, CC);

  dim3 gfg(NN / 64, DD / 16, BB);
  k_proj<<<gfg, 128, 0, stream>>>(xb, wfT, bf_, fb, NN, CC, DD, 0);
  k_proj<<<gfg, 128, 0, stream>>>(xb, wgT, bg,  gb, NN, CC, DD, 0);
  dim3 gh(NN / 64, CC / 16, BB);
  k_proj<<<gh, 128, 0, stream>>>(xb, whT, bh, hTb, NN, CC, CC, 1);

  k_attn<<<dim3(NN / 64, BB), 512, 0, stream>>>(gb, fb, hTb, x, gamma, y);
}
